// Attention_67164698575196
// MI455X (gfx1250) — matrix operand order repaired and checked by reading, not compiled, not run on hardware
//
#include <hip/hip_runtime.h>

// ---------------------------------------------------------------------------
// Show-attend-tell attention for MI455X (gfx1250, wave32, WMMA).
//   GEMM1 att_enc = e @ W_enc + b_enc     (bf16 WMMA, 128x128 tile, dbl-buf)
//   GEMM2 att_dec = dec @ W_dec + b_dec   (bf16 WMMA, 128x128 tile, dbl-buf)
//   scores: fused relu(att_enc+att_dec).w_alpha, att_enc cached in LDS
//   softmax over P
//   GEMM3 att_res = alpha @ att_enc       (bf16 WMMA, 32x64 tile)
// LDS tiles are stored K-contiguous (A[m][k], Bt[n][k], stride 40 ushorts)
// so every WMMA fragment is two ds_load_b128 per lane. Guarded loads are
// clamped-address + cndmask (no EXEC branching); bf16 packing uses native
// f32->bf16 converts.
// ---------------------------------------------------------------------------

typedef __attribute__((ext_vector_type(16))) __bf16         v16bf;
typedef __attribute__((ext_vector_type(16))) unsigned short v16us;
typedef __attribute__((ext_vector_type(8)))  unsigned short v8us;
typedef __attribute__((ext_vector_type(8)))  float          v8f;

#define BB 32
#define CC 2048
#define PP 196
#define AA 512
#define DD 1024
#define GG 20
#define LDK 40   // LDS row stride in ushorts (K=32 padded; 80B=20 dwords -> conflict-free)

__device__ __forceinline__ v8us pack8(const float* v) {
  v8us r;
#pragma unroll
  for (int i = 0; i < 8; ++i)
    r[i] = __builtin_bit_cast(unsigned short, (__bf16)v[i]);  // native RNE cvt
  return r;
}

// A fragment (16x32 bf16 MxK), ISA 7.12.2. elems 0..7 <-> K=khi..khi+7,
// elems 8..15 <-> K=16+khi..16+khi+7 (khi = 8 for lanes 16..31).
__device__ __forceinline__ v16bf fragA_ld(const unsigned short* t, int m0, int lane) {
  const unsigned short* p = t + (m0 + (lane & 15)) * LDK + ((lane & 16) ? 8 : 0);
  v8us lo = *(const v8us*)p;          // 16B aligned: LDK%8==0
  v8us hi = *(const v8us*)(p + 16);
  v16us r;
#pragma unroll
  for (int i = 0; i < 8; ++i) { r[i] = lo[i]; r[8 + i] = hi[i]; }
  return __builtin_bit_cast(v16bf, r);
}

// B fragment (32x16 bf16 KxN) read from TRANSPOSED LDS tile Bt[n][k]:
// lane = (n&15) + 16*khalf, elems 0..7 <-> K = 8*khalf + e, elems 8..15 <-> K = 16 + 8*khalf + (e-8).
__device__ __forceinline__ v16bf fragB_ld(const unsigned short* t, int n0, int lane) {
  const unsigned short* p = t + (n0 + (lane & 15)) * LDK + ((lane & 16) ? 8 : 0);
  v8us lo = *(const v8us*)p;
  v8us hi = *(const v8us*)(p + 16);
  v16us r;
#pragma unroll
  for (int i = 0; i < 8; ++i) { r[i] = lo[i]; r[8 + i] = hi[i]; }
  return __builtin_bit_cast(v16bf, r);
}

// ------------------- GEMM1: att_enc = e @ W_enc + b_enc ---------------------
// e[b,p,c] = enc[b,c,p]; M=196 (2 tiles of 128), K=2048, N=512 (4 tiles)
__global__ __launch_bounds__(256)
void att_enc_gemm(const float* __restrict__ enc, const float* __restrict__ W_enc,
                  const float* __restrict__ b_enc, float* __restrict__ out) {
  __shared__ __align__(16) unsigned short sA[2][128 * LDK];
  __shared__ __align__(16) unsigned short sB[2][128 * LDK];
  const int tid = threadIdx.x, lane = tid & 31, wid = tid >> 5;
  const int wm = (wid >> 1) * 32;            // 4 wave rows
  const int wn = (wid & 1) * 64;             // 2 wave cols
  const int b = blockIdx.z, m0 = blockIdx.y * 128, n0 = blockIdx.x * 128;
  const float* encB = enc + (size_t)b * CC * PP;

  v8us aP[2], bP[2];
  auto loadRegs = [&](int k0) {
#pragma unroll
    for (int s = 0; s < 2; ++s) {
      int t = tid + s * 256, mn = t & 127, kr = t >> 7;
      const int  row  = m0 + mn;
      const bool av   = row < PP;            // loop-invariant per thread
      const int  rowc = av ? row : 0;        // clamped, always in-bounds
      float tmp[8];
#pragma unroll
      for (int j = 0; j < 8; ++j) {          // coalesced over p across lanes
        int k = k0 + kr * 8 + j;
        float x = encB[(size_t)k * PP + rowc];
        tmp[j] = av ? x : 0.f;               // v_cndmask, no exec branch
      }
      aP[s] = pack8(tmp);
#pragma unroll
      for (int j = 0; j < 8; ++j) {          // coalesced over n across lanes
        int k = k0 + kr * 8 + j;
        tmp[j] = W_enc[(size_t)k * AA + n0 + mn];
      }
      bP[s] = pack8(tmp);
    }
  };
  auto storeRegs = [&](int buf) {
#pragma unroll
    for (int s = 0; s < 2; ++s) {
      int t = tid + s * 256, mn = t & 127, kr = t >> 7;
      *(v8us*)&sA[buf][mn * LDK + kr * 8] = aP[s];
      *(v8us*)&sB[buf][mn * LDK + kr * 8] = bP[s];
    }
  };

  v8f acc[2][4];
#pragma unroll
  for (int i = 0; i < 2; ++i)
#pragma unroll
    for (int j = 0; j < 4; ++j) acc[i][j] = (v8f){0.f,0.f,0.f,0.f,0.f,0.f,0.f,0.f};

  loadRegs(0);
  storeRegs(0);
  __syncthreads();
  const int STEPS = CC / 32;
  for (int kt = 0; kt < STEPS; ++kt) {
    int cur = kt & 1;
    if (kt + 1 < STEPS) loadRegs((kt + 1) * 32);
    v16bf af[2], bf[4];
#pragma unroll
    for (int i = 0; i < 2; ++i) af[i] = fragA_ld(sA[cur], wm + i * 16, lane);
#pragma unroll
    for (int j = 0; j < 4; ++j) bf[j] = fragB_ld(sB[cur], wn + j * 16, lane);
#pragma unroll
    for (int i = 0; i < 2; ++i)
#pragma unroll
      for (int j = 0; j < 4; ++j)
        acc[i][j] = __builtin_amdgcn_wmma_f32_16x16x32_bf16(false, af[i], false, bf[j],
                                                            (short)0, acc[i][j], false, false);
    if (kt + 1 < STEPS) storeRegs(cur ^ 1);
    __syncthreads();
  }

  const int colb = n0 + wn + (lane & 15);
  const int rb   = m0 + wm + ((lane & 16) ? 8 : 0);
#pragma unroll
  for (int i = 0; i < 2; ++i)
#pragma unroll
    for (int j = 0; j < 4; ++j) {
      int col = colb + j * 16;
      float bias = b_enc[col];
#pragma unroll
      for (int r = 0; r < 8; ++r) {
        int row = rb + i * 16 + r;
        if (row < PP) out[((size_t)(b * PP + row)) * AA + col] = acc[i][j][r] + bias;
      }
    }
}

// ------------------- GEMM2: att_dec = dec @ W_dec + b_dec -------------------
// flattened M = B*G = 640 = 5x128, K=1024, N=512; dec is row-major.
__global__ __launch_bounds__(256)
void att_dec_gemm(const float* __restrict__ dec, const float* __restrict__ W_dec,
                  const float* __restrict__ b_dec, float* __restrict__ out) {
  __shared__ __align__(16) unsigned short sA[2][128 * LDK];
  __shared__ __align__(16) unsigned short sB[2][128 * LDK];
  const int tid = threadIdx.x, lane = tid & 31, wid = tid >> 5;
  const int wm = (wid >> 1) * 32, wn = (wid & 1) * 64;
  const int m0 = blockIdx.y * 128, n0 = blockIdx.x * 128;

  v8us aP[2], bP[2];
  auto loadRegs = [&](int k0) {
#pragma unroll
    for (int s = 0; s < 2; ++s) {
      int t = tid + s * 256, mn = t & 127, kr = t >> 7;
      float tmp[8];
#pragma unroll
      for (int j = 0; j < 8; ++j)             // per-lane contiguous 32B run
        tmp[j] = dec[(size_t)(m0 + mn) * DD + k0 + kr * 8 + j];
      aP[s] = pack8(tmp);
#pragma unroll
      for (int j = 0; j < 8; ++j)
        tmp[j] = W_dec[(size_t)(k0 + kr * 8 + j) * AA + n0 + mn];
      bP[s] = pack8(tmp);
    }
  };
  auto storeRegs = [&](int buf) {
#pragma unroll
    for (int s = 0; s < 2; ++s) {
      int t = tid + s * 256, mn = t & 127, kr = t >> 7;
      *(v8us*)&sA[buf][mn * LDK + kr * 8] = aP[s];
      *(v8us*)&sB[buf][mn * LDK + kr * 8] = bP[s];
    }
  };

  v8f acc[2][4];
#pragma unroll
  for (int i = 0; i < 2; ++i)
#pragma unroll
    for (int j = 0; j < 4; ++j) acc[i][j] = (v8f){0.f,0.f,0.f,0.f,0.f,0.f,0.f,0.f};

  loadRegs(0);
  storeRegs(0);
  __syncthreads();
  const int STEPS = DD / 32;
  for (int kt = 0; kt < STEPS; ++kt) {
    int cur = kt & 1;
    if (kt + 1 < STEPS) loadRegs((kt + 1) * 32);
    v16bf af[2], bf[4];
#pragma unroll
    for (int i = 0; i < 2; ++i) af[i] = fragA_ld(sA[cur], wm + i * 16, lane);
#pragma unroll
    for (int j = 0; j < 4; ++j) bf[j] = fragB_ld(sB[cur], wn + j * 16, lane);
#pragma unroll
    for (int i = 0; i < 2; ++i)
#pragma unroll
      for (int j = 0; j < 4; ++j)
        acc[i][j] = __builtin_amdgcn_wmma_f32_16x16x32_bf16(false, af[i], false, bf[j],
                                                            (short)0, acc[i][j], false, false);
    if (kt + 1 < STEPS) storeRegs(cur ^ 1);
    __syncthreads();
  }

  const int colb = n0 + wn + (lane & 15);
  const int rb   = m0 + wm + ((lane & 16) ? 8 : 0);
#pragma unroll
  for (int i = 0; i < 2; ++i)
#pragma unroll
    for (int j = 0; j < 4; ++j) {
      int col = colb + j * 16;
      float bias = b_dec[col];
#pragma unroll
      for (int r = 0; r < 8; ++r)
        out[((size_t)(rb + i * 16 + r)) * AA + col] = acc[i][j][r] + bias;
    }
}

// ---- scores: fused relu(att_enc+att_dec) . w_alpha, att_enc cached in LDS --
__global__ __launch_bounds__(256)
void scores_kernel(const float* __restrict__ att_enc, const float* __restrict__ att_dec,
                   const float* __restrict__ w_alpha, const float* __restrict__ b_alpha,
                   float* __restrict__ scores) {
  __shared__ float senc[28 * AA];   // 57 KB
  __shared__ float swa[AA];
  const int pc = blockIdx.x, b = blockIdx.y;
  const int tid = threadIdx.x, lane = tid & 31, wid = tid >> 5;

  for (int e = tid; e < 28 * AA; e += 256)
    senc[e] = att_enc[((size_t)(b * PP + pc * 28 + (e >> 9))) * AA + (e & (AA - 1))];
  for (int a = tid; a < AA; a += 256) swa[a] = w_alpha[a];
  __syncthreads();

  const float balpha = b_alpha[0];
  for (int w = wid; w < GG * 28; w += 8) {        // one wave per (g, p) pair
    int g = w / 28, pl = w - g * 28;
    const float* dptr = att_dec + ((size_t)(b * GG + g)) * AA;
    float sum = 0.f;
#pragma unroll
    for (int j = 0; j < 16; ++j) {
      int a = j * 32 + lane;                      // conflict-free, coalesced
      float x = senc[pl * AA + a] + dptr[a];
      sum += fmaxf(x, 0.f) * swa[a];
    }
    for (int off = 16; off; off >>= 1) sum += __shfl_xor(sum, off, 32);
    if (lane == 0)
      scores[((size_t)(b * GG + g)) * PP + pc * 28 + pl] = sum + balpha;
  }
}

// ------------------------------ softmax over P ------------------------------
__global__ __launch_bounds__(256)
void softmax_kernel(const float* __restrict__ scores, float* __restrict__ alpha) {
  __shared__ float sh[8];
  const int bg = blockIdx.x, tid = threadIdx.x;
  float s = (tid < PP) ? scores[(size_t)bg * PP + tid] : -3.402823466e38f;
  float m = s;
  for (int off = 16; off; off >>= 1) m = fmaxf(m, __shfl_xor(m, off, 32));
  if ((tid & 31) == 0) sh[tid >> 5] = m;
  __syncthreads();
  float bm = sh[0];
#pragma unroll
  for (int w = 1; w < 8; ++w) bm = fmaxf(bm, sh[w]);
  __syncthreads();
  float e = (tid < PP) ? __expf(s - bm) : 0.f;
  float sum = e;
  for (int off = 16; off; off >>= 1) sum += __shfl_xor(sum, off, 32);
  if ((tid & 31) == 0) sh[tid >> 5] = sum;
  __syncthreads();
  float bs = 0.f;
#pragma unroll
  for (int w = 0; w < 8; ++w) bs += sh[w];
  if (tid < PP) alpha[(size_t)bg * PP + tid] = e / bs;
}

// ---------------- GEMM3: att_res = alpha(20x196) @ att_enc(196x512) ---------
// per batch; M padded 20->32, K padded 196->224 with zeros; N tile 64.
__global__ __launch_bounds__(256)
void att_res_gemm(const float* __restrict__ alpha, const float* __restrict__ att_enc,
                  float* __restrict__ out) {
  __shared__ __align__(16) unsigned short sA[32 * LDK];
  __shared__ __align__(16) unsigned short sBt[64 * LDK];
  const int tid = threadIdx.x, lane = tid & 31, wid = tid >> 5;
  const int wm = (wid >> 2) * 16, wn = (wid & 3) * 16;
  const int b = blockIdx.y, n0 = blockIdx.x * 64;
  v8f acc = {0.f, 0.f, 0.f, 0.f, 0.f, 0.f, 0.f, 0.f};

  for (int k0 = 0; k0 < 224; k0 += 32) {
    if (tid < 128) {
      int m = tid & 31, kr = tid >> 5;
      const bool mv = m < GG;
      const int  mc = mv ? m : 0;
      float tmp[8];
#pragma unroll
      for (int j = 0; j < 8; ++j) {
        int  k  = k0 + kr * 8 + j;
        bool kv = k < PP;
        int  kc = kv ? k : 0;
        float x = alpha[((size_t)(b * GG + mc)) * PP + kc];
        tmp[j] = (mv && kv) ? x : 0.f;
      }
      *(v8us*)&sA[m * LDK + kr * 8] = pack8(tmp);
    }
    {
      int n = tid & 63, kr = tid >> 6;
      float tmp[8];
#pragma unroll
      for (int j = 0; j < 8; ++j) {
        int  k  = k0 + kr * 8 + j;
        bool kv = k < PP;
        int  kc = kv ? k : 0;
        float x = att_enc[((size_t)(b * PP + kc)) * AA + n0 + n];
        tmp[j] = kv ? x : 0.f;
      }
      *(v8us*)&sBt[n * LDK + kr * 8] = pack8(tmp);
    }
    __syncthreads();
    v16bf af = fragA_ld(sA, wm, lane);
    v16bf bf = fragB_ld(sBt, wn, lane);
    acc = __builtin_amdgcn_wmma_f32_16x16x32_bf16(false, af, false, bf,
                                                  (short)0, acc, false, false);
    __syncthreads();
  }
  const int col = n0 + wn + (lane & 15);
  const int rb  = wm + ((lane & 16) ? 8 : 0);
#pragma unroll
  for (int r = 0; r < 8; ++r) {
    int g = rb + r;
    if (g < GG) out[((size_t)(b * GG + g)) * AA + col] = acc[r];
  }
}

extern "C" void kernel_launch(void* const* d_in, const int* in_sizes, int n_in,
                              void* d_out, int out_size, void* d_ws, size_t ws_size,
                              hipStream_t stream) {
  (void)in_sizes; (void)n_in; (void)out_size; (void)ws_size;
  const float* enc     = (const float*)d_in[0];
  const float* dec     = (const float*)d_in[1];
  const float* W_enc   = (const float*)d_in[2];
  const float* b_enc   = (const float*)d_in[3];
  const float* W_dec   = (const float*)d_in[4];
  const float* b_dec   = (const float*)d_in[5];
  const float* w_alpha = (const float*)d_in[6];
  const float* b_alpha = (const float*)d_in[7];

  float* out     = (float*)d_out;
  float* ws      = (float*)d_ws;
  float* att_enc = ws;                                    // 32*196*512 f32
  float* att_dec = att_enc + (size_t)BB * PP * AA;        // 32*20*512
  float* scores  = att_dec + (size_t)BB * GG * AA;        // 32*20*196
  float* att_res = out;                                   // output 0
  float* alpha   = out + (size_t)BB * GG * AA;            // output 1

  att_enc_gemm<<<dim3(4, 2, BB), 256, 0, stream>>>(enc, W_enc, b_enc, att_enc);
  att_dec_gemm<<<dim3(4, 5, 1), 256, 0, stream>>>(dec, W_dec, b_dec, att_dec);
  scores_kernel<<<dim3(7, BB), 256, 0, stream>>>(att_enc, att_dec, w_alpha, b_alpha, scores);
  softmax_kernel<<<BB * GG, 256, 0, stream>>>(scores, alpha);
  att_res_gemm<<<dim3(8, BB), 256, 0, stream>>>(alpha, att_enc, att_res);
}
